// PQCompressorBig_64699387347657
// MI455X (gfx1250) — matrix operand order repaired and checked by reading, not compiled, not run on hardware
//
#include <hip/hip_runtime.h>
#include <math.h>

// ---------------------------------------------------------------------------
// CDNA5 (gfx1250) WMMA types: wave32, 16x16x32 bf16 -> f32 accumulate.
// ---------------------------------------------------------------------------
typedef __attribute__((ext_vector_type(16))) __bf16 v16bf;
typedef __attribute__((ext_vector_type(8)))  float  v8f;

// native f32 -> bf16 (v_cvt_*_bf16_f32), no manual rounding chains
__device__ __forceinline__ unsigned short bfbits(float f) {
  return __builtin_bit_cast(unsigned short, (__bf16)f);
}
__device__ __forceinline__ unsigned pk2bf(float a, float b) {
  return (unsigned)bfbits(a) | ((unsigned)bfbits(b) << 16);
}
__device__ __forceinline__ __bf16 f2bf(float f) {
  return (__bf16)f;
}

// 16-bit A-matrix 16x32 layout (ISA 7.12.2): k-pairs are consecutive.
__device__ __forceinline__ int a_kidx(int v, int e, int h) {
  return ((v < 4) ? (2 * v) : (16 + 2 * (v - 4))) + 8 * h + e;
}
// 16-bit B-matrix 32x16 layout: same k order as A (k-pairs, +8 for lanes 16-31).
__device__ __forceinline__ int b_kidx(int v, int e, int h) {
  return ((v < 4) ? (2 * v) : (16 + 2 * (v - 4))) + 8 * h + e;
}

// branch-free im2col fetch (clamp address, select zero)
__device__ __forceinline__ float im2col_val(const float* __restrict__ in,
    int n, int Cin, int Hin, int Win,
    int ci, int ky, int kx, int oy, int ox,
    int stride, int mode, int padc, bool kok)
{
  int iy, ix; bool valid = kok;
  if (mode == 0) {                       // direct conv, SAME
    iy = oy * stride + ky - padc;
    ix = ox * stride + kx - padc;
    valid = valid && ((unsigned)iy < (unsigned)Hin) && ((unsigned)ix < (unsigned)Win);
  } else {                               // conv_transpose s=2, pad_a=2 on dilated input
    int dy = oy + ky - 2, dx = ox + kx - 2;
    valid = valid && (dy >= 0) && (dx >= 0) && !(dy & 1) && !(dx & 1);
    iy = dy >> 1; ix = dx >> 1;
    valid = valid && (iy < Hin) && (ix < Win);
  }
  int iyc = iy < 0 ? 0 : (iy >= Hin ? Hin - 1 : iy);
  int ixc = ix < 0 ? 0 : (ix >= Win ? Win - 1 : ix);
  float v = in[(((long)n * Cin + ci) * Hin + iyc) * Win + ixc];
  return valid ? v : 0.f;
}

// ---------------------------------------------------------------------------
// 3x3 conv (mode 0) / conv_transpose x2 (mode 1), SAME padding, NCHW.
// Block = 128 threads (4 waves) -> 64 Cout x 64 pixel tile (16 WMMA tiles).
// Each wave owns a 16-Cout strip and 4 accumulators; LDS tiles (bf16) are
// reused 4x by every wave -> 4 WMMAs per staging round, ~16 FLOP/B staged.
// All spatial dims are powers of two: wsh=log2(Wout), hwsh=log2(Hout*Wout).
// act: 0=none 1=relu 2=tanh. addsrc: optional fused add.
// ---------------------------------------------------------------------------
__global__ void __launch_bounds__(128)
conv3x3_wmma(const float* __restrict__ in, const float* __restrict__ wgt,
             float* __restrict__ out, const float* __restrict__ addsrc,
             int Nn, int Cin, int Cout, int Hin, int Win,
             int Hout, int Wout, int stride, int mode, int act,
             int wsh, int hwsh)
{
  __shared__ unsigned short sA[64 * 36];   // [co][k], padded row = 36 halves
  __shared__ unsigned short sB[64 * 36];   // [pixel][k]

  const int t    = threadIdx.x;
  const int wave = t >> 5;
  const int lane = t & 31;
  const int half = lane >> 4;
  const int nidx = lane & 15;
  const int cobase = blockIdx.y * 64;
  const long p0 = (long)blockIdx.x * 64;
  const int Ktot = Cin * 9;
  int ptc = (Hout - 1) * stride + 3 - Hin; if (ptc < 0) ptc = 0;
  const int padc = (mode == 0) ? (ptc >> 1) : 2;
  const int wmask = Wout - 1;
  const int n_img = (int)(p0 >> hwsh);     // HW >= 256, tile never spans images

  // B-staging pixel coords (two pixels per lane), shifts/masks only
  const int oyA = ((int)((p0 + lane) >> wsh)) & (Hout - 1);
  const int oxA = ((int)(p0 + lane)) & wmask;
  const int oyB = ((int)((p0 + 32 + lane) >> wsh)) & (Hout - 1);
  const int oxB = ((int)(p0 + 32 + lane)) & wmask;

  // A-staging role: co = t>>1 (64 rows), 16 consecutive k per thread
  const int co_s = t >> 1;
  const int ks0  = (t & 1) * 16;
  int coS = cobase + co_s;
  const bool coOk = coS < Cout;
  if (!coOk) coS = Cout - 1;
  const long wrowS = (long)coS * Ktot;     // weights per co are contiguous

  unsigned*       sA32 = (unsigned*)sA;
  const unsigned* a32  = (const unsigned*)sA;
  const unsigned* b32  = (const unsigned*)sB;
  const int arow18 = (wave * 16 + nidx) * 18;

  v8f acc[4] = {};
  for (int kb = 0; kb < Ktot; kb += 32) {
    // ---- stage A: 64x32 weight tile (packed b32 stores)
    {
      int kg0 = kb + ks0;
      int kpf = kg0 + 32 < Ktot ? kg0 + 32 : Ktot - 1;
      __builtin_prefetch(wgt + wrowS + kpf, 0, 1);       // global_prefetch_b8
#pragma unroll
      for (int j = 0; j < 8; ++j) {
        int k0 = kg0 + 2 * j, k1 = k0 + 1;
        int c0 = k0 < Ktot ? k0 : Ktot - 1;
        int c1 = k1 < Ktot ? k1 : Ktot - 1;
        float w0 = wgt[wrowS + c0]; if (!(coOk && k0 < Ktot)) w0 = 0.f;
        float w1 = wgt[wrowS + c1]; if (!(coOk && k1 < Ktot)) w1 = 0.f;
        sA32[((co_s * 36 + ks0) >> 1) + j] = pk2bf(w0, w1);
      }
    }
    // ---- stage B: 32k x 64px im2col tile (one div-by-9 per staged k)
#pragma unroll
    for (int u = 0; u < 8; ++u) {
      int kl = wave + u * 4;
      int kg = kb + kl;
      bool kok = kg < Ktot;
      int kc = kok ? kg : Ktot - 1;
      int ci = kc / 9; int tt = kc - ci * 9;
      int ky = tt / 3; int kx = tt - ky * 3;
      float vA = im2col_val(in, n_img, Cin, Hin, Win, ci, ky, kx, oyA, oxA, stride, mode, padc, kok);
      float vB = im2col_val(in, n_img, Cin, Hin, Win, ci, ky, kx, oyB, oxB, stride, mode, padc, kok);
      sB[lane * 36 + kl]        = bfbits(vA);
      sB[(lane + 32) * 36 + kl] = bfbits(vB);
    }
    __syncthreads();
    // ---- A fragment once, then 4 B fragments + 4 WMMAs
    union { v16bf v; unsigned u[8]; } fa;
#pragma unroll
    for (int vv = 0; vv < 8; ++vv) {
      int ka = ((vv < 4) ? 2 * vv : 16 + 2 * (vv - 4)) + 8 * half;  // even
      fa.u[vv] = a32[arow18 + (ka >> 1)];
    }
#pragma unroll
    for (int j = 0; j < 4; ++j) {
      union { v16bf v; unsigned u[8]; } fb;
      const int brow18 = (j * 16 + nidx) * 18;
#pragma unroll
      for (int vv = 0; vv < 8; ++vv) {
        int kbp = ((vv < 4) ? 2 * vv : 16 + 2 * (vv - 4)) + 8 * half; // even
        fb.u[vv] = b32[brow18 + (kbp >> 1)];
      }
      acc[j] = __builtin_amdgcn_wmma_f32_16x16x32_bf16(false, fa.v, false, fb.v,
                                                       (short)0, acc[j], false, false);
    }
    __syncthreads();
  }

  // ---- epilogue: C/D layout M = r + 8*half, N = j*16 + nidx
  const int myco0 = cobase + wave * 16;
#pragma unroll
  for (int j = 0; j < 4; ++j) {
    const long p = p0 + j * 16 + nidx;
    const int oy = ((int)(p >> wsh)) & (Hout - 1);
    const int ox = ((int)p) & wmask;
#pragma unroll
    for (int r = 0; r < 8; ++r) {
      int co = myco0 + r + 8 * half;
      if (co >= Cout) continue;
      long oidx = (((long)n_img * Cout + co) * (long)Hout + oy) * Wout + ox;
      float v = acc[j][r];
      if (addsrc) v += addsrc[oidx];
      if (act == 1)      v = v > 0.f ? v : 0.f;
      else if (act == 2) v = tanhf(v);
      out[oidx] = v;
    }
  }
}

// ---------------------------------------------------------------------------
// PQ logits:  l[b,m,yx,k] = 2*<x_g, cb[m][k]> - |x|^2 - |cb|^2
// One wave -> 16 pixels x 16 codes, K(=D)=64 -> two bf16 WMMA steps.
// P and HW are powers of two; hwshl = log2(HW).
// ---------------------------------------------------------------------------
__global__ void pq_logits_wmma(const float* __restrict__ xsrc,  // NCHW, C=256
                               const float* __restrict__ cb,    // (4,2048,64)
                               const float* __restrict__ x2b,   // (4,P)
                               const float* __restrict__ c2b,   // (4,2048)
                               float* __restrict__ lout,        // (B,4,HW,2048)
                               int P, int HW, int hwshl)
{
  const int m    = blockIdx.z;
  const int lane = threadIdx.x & 31;
  const int half = lane >> 4;
  const int nidx = lane & 15;
  const int p0    = blockIdx.x * 16;
  const int code0 = blockIdx.y * 16;
  const int prow = p0 + nidx;                  // A row (pixel)
  const int code = code0 + nidx;               // B col (codeword)
  const int ab  = prow >> hwshl;
  const int ayx = prow & (HW - 1);
  const long xbase = ((long)ab * 256 + m * 64) * HW + ayx;
  const long cbrow = ((long)m * 2048 + code) * 64;

  v8f acc = {};
#pragma unroll
  for (int kb = 0; kb < 64; kb += 32) {
    v16bf va, vb;
#pragma unroll
    for (int v = 0; v < 8; ++v) {
#pragma unroll
      for (int e = 0; e < 2; ++e) {
        int kA = kb + a_kidx(v, e, half);
        va[2 * v + e] = f2bf(xsrc[xbase + (long)kA * HW]);
        int kB = kb + b_kidx(v, e, half);
        vb[2 * v + e] = f2bf(cb[cbrow + kB]);
      }
    }
    acc = __builtin_amdgcn_wmma_f32_16x16x32_bf16(false, va, false, vb,
                                                  (short)0, acc, false, false);
  }
  const float c2v = c2b[m * 2048 + code];
#pragma unroll
  for (int r = 0; r < 8; ++r) {
    int pr = p0 + r + 8 * half;
    int b = pr >> hwshl, yx = pr & (HW - 1);
    float lg = 2.f * acc[r] - x2b[m * P + pr] - c2v;
    lout[((long)(b * 4 + m) * HW + yx) * 2048 + code] = lg;
  }
}

// ---------------------------------------------------------------------------
// |x|^2 per (m, pixel) and |cb|^2 per (m, code)
// ---------------------------------------------------------------------------
__global__ void rowsq_kernel(const float* __restrict__ xsrc,
                             float* __restrict__ x2b, int P, int HW)
{
  int t = blockIdx.x * blockDim.x + threadIdx.x;
  if (t >= 4 * P) return;
  int m = t / P, p = t - m * P;
  int b = p / HW, yx = p - b * HW;
  const float* base = xsrc + ((long)b * 256 + m * 64) * HW + yx;
  float s = 0.f;
  for (int d = 0; d < 64; ++d) { float v = base[(long)d * HW]; s += v * v; }
  x2b[t] = s;
}

__global__ void cbsq_kernel(const float* __restrict__ cb, float* __restrict__ c2b)
{
  int t = blockIdx.x * blockDim.x + threadIdx.x;
  if (t >= 4 * 2048) return;
  const float* r = cb + (long)t * 64;
  float s = 0.f;
  for (int d = 0; d < 64; ++d) s += r[d] * r[d];
  c2b[t] = s;
}

// ---------------------------------------------------------------------------
// Softmax over K=2048 + soft/hard codebook mix + argmax codes.
// One 64-thread block per (b,m,y,x) row; exp row staged in LDS (8KB).
// ---------------------------------------------------------------------------
__global__ void pq_softmax_kernel(const float* __restrict__ lbuf,
                                  const float* __restrict__ cb,
                                  float* __restrict__ soft,   // NCHW C=256
                                  float* __restrict__ hard,   // NCHW C=256
                                  int* __restrict__ codes,    // (B,4,h,w)
                                  const unsigned* __restrict__ temp_raw,
                                  int HW)
{
  __shared__ float sval[64];
  __shared__ int   sidx[64];
  __shared__ float ssum[64];
  __shared__ float se[2048];
  __shared__ float smax_s;
  __shared__ float ssum_s;
  __shared__ int   sarg_s;

  const int tid = threadIdx.x;                 // 0..63
  const int row = blockIdx.x;                  // (b*4+m)*HW + yx
  const int m  = (row / HW) & 3;
  const int b  = row / (4 * HW);
  const int yx = row % HW;
  const float* L = lbuf + (long)row * 2048;

  unsigned tr = *temp_raw;                     // temp may arrive as int or f32
  float tf = __builtin_bit_cast(float, tr);
  float temp = (tf > 1e-30f && tf < 1e30f) ? tf : (float)(int)tr;
  const float invt = 1.f / temp;

  // pass 1: max + first-argmax
  float bm = -3.4e38f; int bi = 0;
  for (int c = tid; c < 2048; c += 64) {
    float v = L[c];
    if (v > bm) { bm = v; bi = c; }
  }
  sval[tid] = bm; sidx[tid] = bi;
  __syncthreads();
  if (tid == 0) {
    float mx = sval[0]; int ix = sidx[0];
    for (int i = 1; i < 64; ++i)
      if (sval[i] > mx || (sval[i] == mx && sidx[i] < ix)) { mx = sval[i]; ix = sidx[i]; }
    smax_s = mx; sarg_s = ix;
  }
  __syncthreads();
  const float mx = smax_s;

  // pass 2: exp row into LDS + sum
  float ls = 0.f;
  for (int c = tid; c < 2048; c += 64) {
    float e = expf((L[c] - mx) * invt);
    se[c] = e; ls += e;
  }
  ssum[tid] = ls;
  __syncthreads();
  if (tid == 0) {
    float s = 0.f;
    for (int i = 0; i < 64; ++i) s += ssum[i];
    ssum_s = s;
  }
  __syncthreads();
  const float inv_sum = 1.f / ssum_s;
  const int argc = sarg_s;

  // pass 3: thread tid == dimension d
  const int d = tid;
  float acc = 0.f;
  const float* cbm = cb + (long)m * 2048 * 64 + d;
  for (int c = 0; c < 2048; ++c) acc += se[c] * cbm[(long)c * 64];
  long oidx = ((long)b * 256 + m * 64 + d) * HW + yx;
  if (soft) soft[oidx] = acc * inv_sum;
  hard[oidx] = cb[((long)m * 2048 + argc) * 64 + d];
  if (tid == 0) codes[row] = argc;
}

__global__ void sub_kernel(const float* __restrict__ a,
                           const float* __restrict__ b,
                           float* __restrict__ o, int n)
{
  int i = blockIdx.x * blockDim.x + threadIdx.x;
  if (i < n) o[i] = a[i] - b[i];
}

// ---------------------------------------------------------------------------
// Host-side orchestration
// ---------------------------------------------------------------------------
extern "C" void kernel_launch(void* const* d_in, const int* in_sizes, int n_in,
                              void* d_out, int out_size, void* d_ws, size_t ws_size,
                              hipStream_t stream)
{
  (void)in_sizes; (void)n_in; (void)out_size; (void)ws_size;
  const float* x      = (const float*)d_in[0];
  const float* enc_w0 = (const float*)d_in[1];
  const float* enc_w1 = (const float*)d_in[2];
  const float* enc_w2 = (const float*)d_in[3];
  const float* enc_w3 = (const float*)d_in[4];
  const float* cb1    = (const float*)d_in[5];
  const float* cb2    = (const float*)d_in[6];
  const float* mq_w   = (const float*)d_in[7];
  const float* mz_w   = (const float*)d_in[8];
  const float* sq_w   = (const float*)d_in[9];
  const float* sz_w   = (const float*)d_in[10];
  const float* dec_w0 = (const float*)d_in[11];
  const float* dec_w1 = (const float*)d_in[12];
  const float* dec_w2 = (const float*)d_in[13];
  const float* dec_w3 = (const float*)d_in[14];
  const float* ctx_w  = (const float*)d_in[15];
  const unsigned* temp_raw = (const unsigned*)d_in[16];

  // outputs (flat, return order): restored, hards, q2, latent, c1, c2, l1, l2, predict
  float* out      = (float*)d_out;
  float* restored = out;
  float* hards    = out + 3145728u;
  float* q2       = out + 4194304u;
  float* latent   = out + 4456448u;
  int*   c1       = (int*)(out + 5505024u);
  int*   c2c      = (int*)(out + 5521408u);
  float* l1       = out + 5525504u;
  float* l2       = out + 39079936u;
  float* predict  = out + 47468544u;

  // workspace (floats); big buffers reused encoder->decoder
  float* ws    = (float*)d_ws;
  float* bufA  = ws;                  // 67,108,864  (h1 / d3)
  float* bufB  = ws + 67108864u;      // 16,777,216  (h2 / d2)
  float* bufC  = ws + 83886080u;      //  4,194,304  (h3 / d1)
  float* softs = ws + 88080384u;      //  1,048,576
  float* q1m   = ws + 89128960u;      //    262,144
  float* zm    = ws + 89391104u;      //    262,144
  float* resid = ws + 89653248u;      //    262,144
  float* rhat  = ws + 89915392u;      //  1,048,576
  float* quant = ws + 90963968u;      //  1,048,576
  float* soft2 = ws + 92012544u;      //    262,144 (unused soft of q2)
  float* x2b   = ws + 92274688u;      //     16,384
  float* c2b   = ws + 92291072u;      //      8,192

  dim3 blk(128);
  dim3 wv(32);

  // ---- encoder: 4x stride-2 conv (relu on first three)
  conv3x3_wmma<<<dim3(4096, 4), blk, 0, stream>>>(x,    enc_w0, bufA,   nullptr, 4, 3,   256, 512, 512, 256, 256, 2, 0, 1, 8, 16);
  conv3x3_wmma<<<dim3(1024, 4), blk, 0, stream>>>(bufA, enc_w1, bufB,   nullptr, 4, 256, 256, 256, 256, 128, 128, 2, 0, 1, 7, 14);
  conv3x3_wmma<<<dim3(256,  4), blk, 0, stream>>>(bufB, enc_w2, bufC,   nullptr, 4, 256, 256, 128, 128, 64,  64,  2, 0, 1, 6, 12);
  conv3x3_wmma<<<dim3(64,   4), blk, 0, stream>>>(bufC, enc_w3, latent, nullptr, 4, 256, 256, 64,  64,  32,  32,  2, 0, 0, 5, 10);

  // ---- quantize #1 (P = 4*32*32 = 4096, HW = 1024)
  rowsq_kernel<<<64, 256, 0, stream>>>(latent, x2b, 4096, 1024);
  cbsq_kernel<<<32, 256, 0, stream>>>(cb1, c2b);
  pq_logits_wmma<<<dim3(256, 128, 4), wv, 0, stream>>>(latent, cb1, x2b, c2b, l1, 4096, 1024, 10);
  pq_softmax_kernel<<<16384, 64, 0, stream>>>(l1, cb1, softs, hards, c1, temp_raw, 1024);

  // ---- residual path
  conv3x3_wmma<<<dim3(16, 4), blk, 0, stream>>>(softs,  mq_w, q1m, nullptr, 4, 256, 256, 32, 32, 16, 16, 2, 0, 0, 4, 8);
  conv3x3_wmma<<<dim3(16, 4), blk, 0, stream>>>(latent, mz_w, zm,  nullptr, 4, 256, 256, 32, 32, 16, 16, 2, 0, 0, 4, 8);
  sub_kernel<<<1024, 256, 0, stream>>>(zm, q1m, resid, 262144);

  // ---- quantize #2 (P = 4*16*16 = 1024, HW = 256)
  rowsq_kernel<<<16, 256, 0, stream>>>(resid, x2b, 1024, 256);
  cbsq_kernel<<<32, 256, 0, stream>>>(cb2, c2b);
  pq_logits_wmma<<<dim3(64, 128, 4), wv, 0, stream>>>(resid, cb2, x2b, c2b, l2, 1024, 256, 8);
  pq_softmax_kernel<<<4096, 64, 0, stream>>>(l2, cb2, soft2, q2, c2c, temp_raw, 256);

  // ---- synthesis: rHat = tconv(q2, sq), quantized = conv(hards, sz) + rHat
  conv3x3_wmma<<<dim3(64, 4), blk, 0, stream>>>(q2,    sq_w, rhat,  nullptr, 4, 256, 256, 16, 16, 32, 32, 2, 1, 0, 5, 10);
  conv3x3_wmma<<<dim3(64, 4), blk, 0, stream>>>(hards, sz_w, quant, rhat,    4, 256, 256, 32, 32, 32, 32, 1, 0, 0, 5, 10);

  // ---- context model: conv(q2, ctx_w), Cout = M*K = 8192
  conv3x3_wmma<<<dim3(16, 128), blk, 0, stream>>>(q2, ctx_w, predict, nullptr, 4, 256, 8192, 16, 16, 16, 16, 1, 0, 0, 4, 8);

  // ---- decoder: 4x tconv (relu, relu, relu, tanh)
  conv3x3_wmma<<<dim3(256,  4), blk, 0, stream>>>(quant, dec_w0, bufC,     nullptr, 4, 256, 256, 32,  32,  64,  64,  2, 1, 1, 6, 12);
  conv3x3_wmma<<<dim3(1024, 4), blk, 0, stream>>>(bufC,  dec_w1, bufB,     nullptr, 4, 256, 256, 64,  64,  128, 128, 2, 1, 1, 7, 14);
  conv3x3_wmma<<<dim3(4096, 4), blk, 0, stream>>>(bufB,  dec_w2, bufA,     nullptr, 4, 256, 256, 128, 128, 256, 256, 2, 1, 1, 8, 16);
  conv3x3_wmma<<<dim3(16384, 1), blk, 0, stream>>>(bufA,  dec_w3, restored, nullptr, 4, 256, 3,   256, 256, 512, 512, 2, 1, 2, 9, 18);
}
